// LocalSlidingWindowAttention_9019431321714
// MI455X (gfx1250) — matrix operand order repaired and checked by reading, not compiled, not run on hardware
//
#include <hip/hip_runtime.h>

// ---------------- problem constants ----------------
constexpr int BATCH  = 4;
constexpr int SEQ    = 4096;
constexpr int EMB    = 1024;
constexpr int NHEADS = 16;
constexpr int HDIM   = 64;      // head dim
constexpr int WSZ    = 128;     // sliding window
constexpr int NBLK   = SEQ / WSZ;   // 32 query blocks

// ---------------- WMMA types ----------------
typedef __attribute__((ext_vector_type(16))) __bf16 v16bf;
typedef __attribute__((ext_vector_type(8)))  __bf16 v8bf;
typedef __attribute__((ext_vector_type(8)))  float  v8f;

static __device__ inline v16bf mk16(v8bf lo, v8bf hi) {
    v16bf r;
#pragma unroll
    for (int i = 0; i < 8; ++i) { r[i] = lo[i]; r[i + 8] = hi[i]; }
    return r;
}

static __device__ inline v8f wmma_bf16(v16bf a, v16bf b, v8f c) {
    // (neg_a, A, neg_b, B, c_mod, C, reuse_a, reuse_b)
    return __builtin_amdgcn_wmma_f32_16x16x32_bf16(false, a, false, b, (short)0, c, false, false);
}

// =====================================================================
// GEMM: Y[M,N] = X[M,K] @ W[K,N] + bias, bf16 WMMA with f32 accumulate.
// MODE 0: store f32       (final out-projection)
// MODE 1: store bf16      (V projection)
// MODE 2: store bf16 with fused RoPE (+ scale)   (Q/K projections)
// Block tile 128x128, K-step 32, 256 threads = 8 waves (wave32),
// double-buffered LDS staging (ping-pong) to overlap loads with WMMA.
// =====================================================================
template<int MODE>
__global__ __launch_bounds__(256) void gemm_kernel(
    const float* __restrict__ X, const float* __restrict__ W,
    const float* __restrict__ bias,
    float* __restrict__ Yf, __bf16* __restrict__ Yb,
    float rope_scale, int M, int N, int K)
{
    __shared__ __align__(16) __bf16 As[2][128][32];   // 16 KB (A tile, bf16)
    __shared__ __align__(16) __bf16 Bt[2][128][32];   // 16 KB (W tile, transposed: Bt[n][k])

    const int tid  = threadIdx.x;
    const int wave = tid >> 5, lane = tid & 31;
    const int ln = lane & 15, lh = lane >> 4;
    const int kbA = lh * 8;     // A-frag: K = {kbA..kbA+7, 16+kbA..16+kbA+7}
    const int kbB = lh * 16;    // B-frag: K = {kbA..kbA+7, 16+kbA..16+kbA+7} (same as A)
    const int m0 = blockIdx.y * 128, n0 = blockIdx.x * 128;
    const int wm = (wave & 3) * 32;   // wave row offset (2 row-tiles)
    const int wn = (wave >> 2) * 64;  // wave col offset (4 col-tiles)

    // staging thread mapping
    const int ar = tid >> 1, ac0 = (tid & 1) * 16;    // A: row, col base
    const int bk = tid & 31, bn0 = (tid >> 5) * 16;   // B: k, n base

    v8f acc[2][4];
#pragma unroll
    for (int i = 0; i < 2; ++i)
#pragma unroll
        for (int j = 0; j < 4; ++j) acc[i][j] = v8f{};

    // ---- prologue: stage tile 0 into buffer 0 ----
    {
        const float* asrc = X + (size_t)(m0 + ar) * K + ac0;
        const float* bsrc = W + (size_t)bk * N + n0 + bn0;
#pragma unroll
        for (int j = 0; j < 16; ++j) As[0][ar][ac0 + j] = (__bf16)asrc[j];
#pragma unroll
        for (int j = 0; j < 16; ++j) Bt[0][bn0 + j][bk] = (__bf16)bsrc[j];
    }
    __syncthreads();

    for (int k0 = 0; k0 < K; k0 += 32) {
        const int cur = (k0 >> 5) & 1;
        const int nxt = cur ^ 1;
        const bool more = (k0 + 32) < K;

        // ---- issue next tile's global loads into registers first ----
        float4 areg[4], breg[4];
        if (more) {
            const float4* asrc = (const float4*)(X + (size_t)(m0 + ar) * K + (k0 + 32) + ac0);
            const float4* bsrc = (const float4*)(W + (size_t)(k0 + 32 + bk) * N + n0 + bn0);
#pragma unroll
            for (int j = 0; j < 4; ++j) areg[j] = asrc[j];
#pragma unroll
            for (int j = 0; j < 4; ++j) breg[j] = bsrc[j];
        }

        // ---- fragments from current buffer + WMMA ----
        v16bf af[2], bf[4];
#pragma unroll
        for (int i = 0; i < 2; ++i) {
            const __bf16* p = &As[cur][wm + i * 16 + ln][0];
            af[i] = mk16(*(const v8bf*)(p + kbA), *(const v8bf*)(p + 16 + kbA));
        }
#pragma unroll
        for (int j = 0; j < 4; ++j) {
            const __bf16* p = &Bt[cur][wn + j * 16 + ln][0];
            bf[j] = mk16(*(const v8bf*)(p + kbA), *(const v8bf*)(p + 16 + kbA));
        }
#pragma unroll
        for (int i = 0; i < 2; ++i)
#pragma unroll
            for (int j = 0; j < 4; ++j)
                acc[i][j] = wmma_bf16(af[i], bf[j], acc[i][j]);

        // ---- convert + store next tile into the other buffer ----
        if (more) {
            const float* af32 = (const float*)areg;
            const float* bf32 = (const float*)breg;
#pragma unroll
            for (int j = 0; j < 16; ++j) As[nxt][ar][ac0 + j] = (__bf16)af32[j];
#pragma unroll
            for (int j = 0; j < 16; ++j) Bt[nxt][bn0 + j][bk] = (__bf16)bf32[j];
        }
        __syncthreads();
    }

    // ---------------- epilogue ----------------
    // C layout: element e -> row = e + 8*lh, col = ln (per 16x16 tile)
#pragma unroll
    for (int i = 0; i < 2; ++i) {
        if constexpr (MODE == 2) {
            // RoPE: head = 64 cols; pair (idx, idx+32) lives in tiles (j, j+2).
#pragma unroll
            for (int j = 0; j < 2; ++j) {
                int col1 = n0 + wn + j * 16 + ln;   // idx = col1 % 64 in [0,32)
                int col2 = col1 + 32;
                int idx  = col1 & 63;
                float invf = powf(10000.0f, -(float)idx * (1.0f / 32.0f));
                float b1 = bias[col1], b2 = bias[col2];
#pragma unroll
                for (int e = 0; e < 8; ++e) {
                    int row = m0 + wm + i * 16 + e + lh * 8;
                    int s   = row & (SEQ - 1);        // seq position
                    float x1 = acc[i][j][e] + b1;
                    float x2 = acc[i][j + 2][e] + b2;
                    float ang = (float)s * invf;
                    float sn, cn;
                    sincosf(ang, &sn, &cn);
                    Yb[(size_t)row * N + col1] = (__bf16)((x1 * cn - x2 * sn) * rope_scale);
                    Yb[(size_t)row * N + col2] = (__bf16)((x1 * sn + x2 * cn) * rope_scale);
                }
            }
        } else {
#pragma unroll
            for (int j = 0; j < 4; ++j) {
                int col = n0 + wn + j * 16 + ln;
                float bb = bias[col];
#pragma unroll
                for (int e = 0; e < 8; ++e) {
                    int row = m0 + wm + i * 16 + e + lh * 8;
                    float v = acc[i][j][e] + bb;
                    if constexpr (MODE == 1) Yb[(size_t)row * N + col] = (__bf16)v;
                    else                     Yf[(size_t)row * N + col] = v;
                }
            }
        }
    }
}

// =====================================================================
// Sliding-window attention. One workgroup per (b, h, 128-query block).
// Keys/values: previous block + current block = 256 positions.
// V tile staged raw into LDS with async global->LDS copies (ASYNCcnt),
// then transposed LDS->LDS. 8 waves; each wave owns 16 query rows.
// Scores 16x256 in registers, softmax via half-wave shfl reductions,
// P staged 16x32 through LDS (reusing the raw-V buffer) for P@V WMMAs.
// =====================================================================
union AttnScratch {
    __bf16 vs[2 * WSZ][HDIM];   // raw V tile (32 KB), dead after transpose
    __bf16 pc[8][16][32];       // per-wave P staging (8 KB)
};

__global__ __launch_bounds__(256) void attn_kernel(
    const __bf16* __restrict__ Qb, const __bf16* __restrict__ Kb,
    const __bf16* __restrict__ Vb, const unsigned char* __restrict__ kpm,
    float* __restrict__ ctx)
{
    __shared__ __align__(16) __bf16 Vt[HDIM][2 * WSZ];  // 32 KB, Vt[d][key]
    __shared__ __align__(16) AttnScratch u;             // 32 KB

    const int blk = blockIdx.x;
    const int n = blk & (NBLK - 1);
    const int h = (blk >> 5) & (NHEADS - 1);
    const int b = blk >> 9;
    const int tid  = threadIdx.x;
    const int wave = tid >> 5, lane = tid & 31;
    const int ln = lane & 15, lh = lane >> 4;
    const int kbA = lh * 8, kbB = lh * 16;

    // ---- async-stage raw V tile: u.vs[key][d] = V[key][d] ----
    {
        int j = tid;                       // key slot 0..255
        int kpos = n * WSZ - WSZ + j;      // global key position
        if (kpos >= 0) {
            const __bf16* src = Vb + (((size_t)b * SEQ + kpos) * EMB + h * HDIM);
            unsigned dst = (unsigned)(unsigned long long)&u.vs[j][0];
#pragma unroll
            for (int c = 0; c < 8; ++c) {   // 64 bf16 = 8 x 16B chunks
                asm volatile("global_load_async_to_lds_b128 %0, %1, off"
                             :: "v"(dst + c * 16), "v"(src + c * 8)
                             : "memory");
            }
        }
        asm volatile("s_wait_asynccnt 0" ::: "memory");
    }
    __syncthreads();

    // ---- transpose LDS->LDS: Vt[d][key] (zero for the left pad block) ----
    {
        int j = tid;
        int kpos = n * WSZ - WSZ + j;
        if (kpos >= 0) {
#pragma unroll
            for (int d = 0; d < HDIM; ++d) Vt[d][j] = u.vs[j][d];
        } else {
#pragma unroll
            for (int d = 0; d < HDIM; ++d) Vt[d][j] = (__bf16)0.0f;
        }
    }
    __syncthreads();

    // ---- Q A-fragments for this wave's 16 rows (head dim = 2 k-steps) ----
    const int qrow = wave * 16 + ln;            // A-frag row = lane&15
    const int qg   = n * WSZ + qrow;
    const __bf16* qsrc = Qb + (((size_t)b * SEQ + qg) * EMB + h * HDIM);
    v16bf qa[2];
#pragma unroll
    for (int d = 0; d < 2; ++d) {
        v8bf lo = *(const v8bf*)(qsrc + d * 32 + kbA);
        v8bf hi = *(const v8bf*)(qsrc + d * 32 + 16 + kbA);
        qa[d] = mk16(lo, hi);
    }

    // ---- scores: 16 tiles of 16x16 over 256 keys ----
    const float NEG = -1e30f;
    v8f sc[16];
#pragma unroll
    for (int t = 0; t < 16; ++t) {
        bool tile_valid = (n > 0) || (t >= 8);   // left-pad tiles are uniformly invalid
        if (tile_valid) {
            int kg = n * WSZ - WSZ + t * 16 + ln;   // B-frag col = lane&15
            const __bf16* ksrc = Kb + (((size_t)b * SEQ + kg) * EMB + h * HDIM);
            v8f a = v8f{};
#pragma unroll
            for (int d = 0; d < 2; ++d) {
                v8bf lo = *(const v8bf*)(ksrc + d * 32 + kbA);
                v8bf hi = *(const v8bf*)(ksrc + d * 32 + 16 + kbA);
                a = wmma_bf16(qa[d], mk16(lo, hi), a);
            }
            unsigned char pad = kpm[(size_t)b * SEQ + kg];
#pragma unroll
            for (int e = 0; e < 8; ++e) {
                int r  = wave * 16 + e + lh * 8;     // local query row of element e
                int qq = n * WSZ + r;
                bool ok = (kg >= qq - WSZ) && (kg <= qq) && (pad == 0);
                a[e] = ok ? a[e] : NEG;
            }
            sc[t] = a;
        } else {
            v8f a;
#pragma unroll
            for (int e = 0; e < 8; ++e) a[e] = NEG;
            sc[t] = a;
        }
    }

    // ---- row softmax (256 cols = 16 tiles x 16 lanes of a half-wave) ----
    float rmax[8], rsum[8], rinv[8];
#pragma unroll
    for (int e = 0; e < 8; ++e) {
        float m = sc[0][e];
#pragma unroll
        for (int t = 1; t < 16; ++t) m = fmaxf(m, sc[t][e]);
#pragma unroll
        for (int off = 1; off < 16; off <<= 1) m = fmaxf(m, __shfl_xor(m, off, 32));
        rmax[e] = m;
        rsum[e] = 0.0f;
    }
#pragma unroll
    for (int t = 0; t < 16; ++t)
#pragma unroll
        for (int e = 0; e < 8; ++e) {
            float p = __expf(sc[t][e] - rmax[e]);
            sc[t][e] = p;
            rsum[e] += p;
        }
#pragma unroll
    for (int e = 0; e < 8; ++e) {
        float s = rsum[e];
#pragma unroll
        for (int off = 1; off < 16; off <<= 1) s += __shfl_xor(s, off, 32);
        rinv[e] = 1.0f / s;
    }

    // ---- P @ V : 8 key-chunks of 32, P staged through per-wave LDS ----
    v8f oacc[4];
#pragma unroll
    for (int dt = 0; dt < 4; ++dt) oacc[dt] = v8f{};
#pragma unroll
    for (int kt = 0; kt < 8; ++kt) {
        // write normalized P chunk (16 rows x 32 keys), bf16
#pragma unroll
        for (int tt = 0; tt < 2; ++tt) {
            int t = kt * 2 + tt;
#pragma unroll
            for (int e = 0; e < 8; ++e)
                u.pc[wave][e + lh * 8][tt * 16 + ln] = (__bf16)(sc[t][e] * rinv[e]);
        }
        // in-wave DS ordering: drain stores before fragment reads
        asm volatile("s_wait_dscnt 0" ::: "memory");
        const __bf16* pr = &u.pc[wave][ln][0];
        v16bf pa = mk16(*(const v8bf*)(pr + kbA), *(const v8bf*)(pr + 16 + kbA));
#pragma unroll
        for (int dt = 0; dt < 4; ++dt) {
            const __bf16* vr = &Vt[dt * 16 + ln][kt * 32 + kbA];
            oacc[dt] = wmma_bf16(pa, mk16(*(const v8bf*)vr, *(const v8bf*)(vr + 16)), oacc[dt]);
        }
    }

    // ---- store ctx (f32, (B,S,D) layout) ----
#pragma unroll
    for (int dt = 0; dt < 4; ++dt)
#pragma unroll
        for (int e = 0; e < 8; ++e) {
            int r = wave * 16 + e + lh * 8;
            ctx[(((size_t)b * SEQ + n * WSZ + r) * EMB + h * HDIM) + dt * 16 + ln] = oacc[dt][e];
        }
}

// =====================================================================
extern "C" void kernel_launch(void* const* d_in, const int* in_sizes, int n_in,
                              void* d_out, int out_size, void* d_ws, size_t ws_size,
                              hipStream_t stream) {
    const float* x  = (const float*)d_in[0];
    const unsigned char* kpm = (const unsigned char*)d_in[1];   // bool mask
    const float* Wq = (const float*)d_in[2];
    const float* bq = (const float*)d_in[3];
    const float* Wk = (const float*)d_in[4];
    const float* bk = (const float*)d_in[5];
    const float* Wv = (const float*)d_in[6];
    const float* bv = (const float*)d_in[7];
    const float* Wo = (const float*)d_in[8];
    const float* bo = (const float*)d_in[9];

    // workspace: Qb/Kb/Vb bf16 (32 MB each) + ctx f32 (64 MB) = 160 MB
    char* ws = (char*)d_ws;
    const size_t BF = (size_t)BATCH * SEQ * EMB * sizeof(__bf16);   // 32 MB
    __bf16* Qb = (__bf16*)(ws);
    __bf16* Kb = (__bf16*)(ws + BF);
    __bf16* Vb = (__bf16*)(ws + 2 * BF);
    float*  ctx = (float*)(ws + 3 * BF);
    float*  out = (float*)d_out;

    const int M = BATCH * SEQ;   // 16384
    dim3 gg(EMB / 128, M / 128); // 8 x 128 tiles
    // Q/K with fused RoPE (+ q scale 1/sqrt(64)), V plain bf16
    gemm_kernel<2><<<gg, 256, 0, stream>>>(x, Wq, bq, nullptr, Qb, 0.125f, M, EMB, EMB);
    gemm_kernel<2><<<gg, 256, 0, stream>>>(x, Wk, bk, nullptr, Kb, 1.0f,   M, EMB, EMB);
    gemm_kernel<1><<<gg, 256, 0, stream>>>(x, Wv, bv, nullptr, Vb, 1.0f,   M, EMB, EMB);
    // attention: (b, h, query-block) grid
    attn_kernel<<<BATCH * NHEADS * NBLK, 256, 0, stream>>>(Qb, Kb, Vb, kpm, ctx);
    // output projection, f32 result
    gemm_kernel<0><<<gg, 256, 0, stream>>>(ctx, Wo, bo, out, nullptr, 1.0f, M, EMB, EMB);
}
